// TransformerBlockQuantum_65481071398466
// MI455X (gfx1250) — matrix operand order repaired and checked by reading, not compiled, not run on hardware
//
#include <hip/hip_runtime.h>
#include <math.h>

typedef __attribute__((ext_vector_type(16))) _Float16 v16h;
typedef __attribute__((ext_vector_type(8)))  float    v8f;

#define NB   8
#define SEQ  2048
#define EMB  8
#define NQC  8
#define FF   4096
#define ROWS (NB * SEQ)          // 16384
#define LNEPS 1e-5f
// (1/sqrt(8)) * log2(e): folded into Qh so softmax is a bare v_exp_f32 (exp2)
#define QSCALE 0.5101061178208458f

#if __has_builtin(__builtin_amdgcn_exp2f)
#define EXP2F(x) __builtin_amdgcn_exp2f(x)
#else
#define EXP2F(x) exp2f(x)
#endif

// WMMA fragment viewed as two 16-byte chunks for vector LDS/global access
union Frag { v16h v; uint4 u[2]; };
union H2  { _Float16 h[2]; unsigned u; };
union H8  { _Float16 h[8]; uint4 u; };
union F8  { float f[8]; float4 v[2]; };

// ---------- helpers ----------

// RX-chain closed form: c = cos(z+th); out[k]=prod_{i<=k} c_i (k>=1), out[0]=prod_{i>=1} c_i
static __device__ inline void qproj_row(const float* z, const float* __restrict__ th, float* out) {
  float c[NQC];
#pragma unroll
  for (int i = 0; i < NQC; ++i) c[i] = __cosf(z[i] + th[i]);
  float tail = 1.f;
#pragma unroll
  for (int i = 1; i < NQC; ++i) tail *= c[i];
  out[0] = tail;
  float run = c[0];
#pragma unroll
  for (int k = 1; k < NQC; ++k) { run *= c[k]; out[k] = run; }
}

// ---------- kernel 1: shared pre-projection + quantum Q/K/V (f16, Q pre-scaled) ----------

__global__ void qkv_kernel(const float* __restrict__ x, const float* __restrict__ w_pre,
                           const float* __restrict__ b_pre, const float* __restrict__ thq,
                           const float* __restrict__ thk, const float* __restrict__ thv,
                           _Float16* __restrict__ Qh, _Float16* __restrict__ Kh,
                           _Float16* __restrict__ Vh) {
  int row = blockIdx.x * blockDim.x + threadIdx.x;
  if (row >= ROWS) return;
  F8 xr;
  xr.v[0] = *(const float4*)(x + row * EMB);
  xr.v[1] = *(const float4*)(x + row * EMB + 4);
  float qin[NQC], o[NQC];
#pragma unroll
  for (int j = 0; j < NQC; ++j) {
    float s = b_pre[j];
#pragma unroll
    for (int i = 0; i < EMB; ++i) s += xr.f[i] * w_pre[i * NQC + j];
    qin[j] = s;
  }
  H8 w;
  qproj_row(qin, thq, o);
#pragma unroll
  for (int j = 0; j < NQC; ++j) w.h[j] = (_Float16)(o[j] * QSCALE);
  *(uint4*)(Qh + row * NQC) = w.u;
  qproj_row(qin, thk, o);
#pragma unroll
  for (int j = 0; j < NQC; ++j) w.h[j] = (_Float16)o[j];
  *(uint4*)(Kh + row * NQC) = w.u;
  qproj_row(qin, thv, o);
#pragma unroll
  for (int j = 0; j < NQC; ++j) w.h[j] = (_Float16)o[j];
  *(uint4*)(Vh + row * NQC) = w.u;
}

// ---------- kernel 1b: pre-pack FFN weights ----------
// w1h[c][0..7] = w1[:,c], w1h[c][8] = b1[c] (bias via A ones slot), rest 0.
// w2p chunk-major with interleaved K: w2p[c/32][n][2*(c%16) + (c%32)/16] = w2[c][n] (n<8).

__global__ void pack_w_kernel(const float* __restrict__ w1, const float* __restrict__ b1,
                              const float* __restrict__ w2,
                              _Float16* __restrict__ w1h, _Float16* __restrict__ w2p) {
  int c = blockIdx.x * blockDim.x + threadIdx.x;
  if (c >= FF) return;
#pragma unroll
  for (int kd = 0; kd < 8; ++kd) w1h[c * 32 + kd] = (_Float16)w1[kd * FF + c];
  w1h[c * 32 + 8] = (_Float16)b1[c];
#pragma unroll
  for (int kd = 9; kd < 32; ++kd) w1h[c * 32 + kd] = (_Float16)0.f;
  int chunk = c >> 5;
  int idx = 2 * (c & 15) + ((c >> 4) & 1);   // interleaved K within the 32-chunk
#pragma unroll
  for (int nn = 0; nn < 16; ++nn)
    w2p[(size_t)chunk * 512 + nn * 32 + idx] = (nn < 8) ? (_Float16)w2[c * 8 + nn] : (_Float16)0.f;
}

// ---------- kernel 2: flash attention, one wave per 16-query tile ----------
// |score| bounded (cos-product inputs): no max subtraction needed. Row-sum l rides the
// P@V WMMA via a ones-row in V's padding. K axis of the P@V GEMM uses an interleaved
// permutation so each (p0,p1) pair is one packed b32 LDS store.

__global__ __launch_bounds__(32) void attn_flash_kernel(
    const _Float16* __restrict__ Qh, const _Float16* __restrict__ Kh,
    const _Float16* __restrict__ Vh, float* __restrict__ attn) {
  const int lane = threadIdx.x;
  const int n = lane & 15;      // B/C-layout column; also A-layout row (M)
  const int grp = lane >> 4;
  const int batch = blockIdx.x >> 7;   // SEQ/16 == 128 q-tiles per batch
  const int qt = blockIdx.x & 127;
  const _Float16* Qb = Qh + (size_t)batch * SEQ * EMB;
  const _Float16* Kb = Kh + (size_t)batch * SEQ * EMB;
  const _Float16* Vb = Vh + (size_t)batch * SEQ * EMB;

  __shared__ __align__(16) _Float16 sQ[16 * 32];
  __shared__ __align__(16) _Float16 sK[32 * 32];
  __shared__ __align__(16) _Float16 sVt[16 * 32];
  __shared__ __align__(16) _Float16 sP[16 * 32];
  __shared__ float sL[16];

  const uint4 z4 = {0, 0, 0, 0};
  {  // zero the zero-padded regions once (sP is fully rewritten each tile)
    uint4* p0 = (uint4*)sQ;   // 64 uint4
    uint4* p1 = (uint4*)sK;   // 128
    uint4* p2 = (uint4*)sVt;  // 64
    p0[lane] = z4; p0[lane + 32] = z4;
    p1[lane] = z4; p1[lane + 32] = z4; p1[lane + 64] = z4; p1[lane + 96] = z4;
    p2[lane] = z4; p2[lane + 32] = z4;
  }
  __syncthreads();
  sVt[8 * 32 + lane] = (_Float16)1.0f;  // ones row: O column 8 accumulates row-sum l
  if (lane < 16)
    *(uint4*)(sQ + lane * 32) = *(const uint4*)(Qb + (qt * 16 + lane) * EMB);
  __syncthreads();

  Frag aQ;
  aQ.u[0] = *(const uint4*)(sQ + n * 32 + grp * 8);  // K 0..15 region
  aQ.u[1] = z4;                                      // K 16..31 is zero by construction

  v8f Oacc = {};
  const int vidx = 2 * n + grp;  // interleaved key slot for this lane's staged V row

#pragma unroll 2
  for (int kt = 0; kt < SEQ / 32; ++kt) {
    const int kbase = kt * 32;
    // stage K tile rows (coalesced b128) + V tile transposed into interleaved K order
    *(uint4*)(sK + lane * 32) = *(const uint4*)(Kb + (kbase + lane) * EMB);
    H8 vr; vr.u = *(const uint4*)(Vb + (kbase + lane) * EMB);
#pragma unroll
    for (int d = 0; d < 8; ++d) sVt[d * 32 + vidx] = vr.h[d];
    if (kbase + 32 < SEQ) {
      __builtin_prefetch(Kb + (kbase + 32 + lane) * EMB, 0, 0);
      __builtin_prefetch(Vb + (kbase + 32 + lane) * EMB, 0, 0);
    }
    __syncthreads();

    Frag bK0, bK1, bV;
    const _Float16* k0p = sK + n * 32 + grp * 8;
    bK0.u[0] = *(const uint4*)k0p;  bK0.u[1] = *(const uint4*)(k0p + 16);
    const _Float16* k1p = sK + (16 + n) * 32 + grp * 8;
    bK1.u[0] = *(const uint4*)k1p;  bK1.u[1] = *(const uint4*)(k1p + 16);
    const _Float16* vp = sVt + n * 32 + grp * 8;
    bV.u[0] = *(const uint4*)vp;    bV.u[1] = *(const uint4*)(vp + 16);

    v8f zc = {};
    v8f S0 = __builtin_amdgcn_wmma_f32_16x16x32_f16(false, aQ.v, false, bK0.v, (short)0, zc, false, false);
    v8f S1 = __builtin_amdgcn_wmma_f32_16x16x32_f16(false, aQ.v, false, bK1.v, (short)0, zc, false, false);

#pragma unroll
    for (int j = 0; j < 8; ++j) {   // p = exp2(S) = e^{QK/sqrt(8)} (scale pre-folded)
      int r = j + 8 * grp;
      H2 pk;
      pk.h[0] = (_Float16)EXP2F(S0[j]);   // key n      -> slot 2n
      pk.h[1] = (_Float16)EXP2F(S1[j]);   // key 16+n   -> slot 2n+1
      *(unsigned*)(sP + r * 32 + 2 * n) = pk.u;
    }
    __syncthreads();

    Frag aP;
    aP.u[0] = *(const uint4*)(sP + n * 32 + grp * 8);
    aP.u[1] = *(const uint4*)(sP + n * 32 + 16 + grp * 8);
    Oacc = __builtin_amdgcn_wmma_f32_16x16x32_f16(false, aP.v, false, bV.v, (short)0, Oacc, false, false);
    __syncthreads();
  }

  if (n == 8) {  // broadcast row-sums (accumulated in C column 8) via LDS
#pragma unroll
    for (int j = 0; j < 8; ++j) sL[j + 8 * grp] = Oacc[j];
  }
  __syncthreads();
#pragma unroll
  for (int j = 0; j < 8; ++j) {
    int r = j + 8 * grp;
    if (n < EMB)
      attn[((size_t)batch * SEQ + qt * 16 + r) * EMB + n] = Oacc[j] / sL[r];
  }
}

// ---------- kernel 3: attn @ w_comb + b_comb, residual, LN1, FFN pre-proj + qproj ----------

__global__ void postattn_kernel(const float* __restrict__ x, const float* __restrict__ attn,
                                const float* __restrict__ w_comb, const float* __restrict__ b_comb,
                                const float* __restrict__ g1, const float* __restrict__ be1,
                                const float* __restrict__ w_pre_f, const float* __restrict__ b_pre_f,
                                const float* __restrict__ th_f,
                                float* __restrict__ Hout, _Float16* __restrict__ QFh) {
  int row = blockIdx.x * blockDim.x + threadIdx.x;
  if (row >= ROWS) return;
  F8 ar, xr, zr;
  ar.v[0] = *(const float4*)(attn + row * EMB);
  ar.v[1] = *(const float4*)(attn + row * EMB + 4);
  xr.v[0] = *(const float4*)(x + row * EMB);
  xr.v[1] = *(const float4*)(x + row * EMB + 4);
#pragma unroll
  for (int j = 0; j < EMB; ++j) {
    float s = b_comb[j];
#pragma unroll
    for (int i = 0; i < EMB; ++i) s += ar.f[i] * w_comb[i * EMB + j];
    zr.f[j] = xr.f[j] + s;
  }
  float mu = 0.f;
#pragma unroll
  for (int j = 0; j < EMB; ++j) mu += zr.f[j];
  mu *= (1.f / EMB);
  float var = 0.f;
#pragma unroll
  for (int j = 0; j < EMB; ++j) { float d = zr.f[j] - mu; var += d * d; }
  var *= (1.f / EMB);
  float rs = rsqrtf(var + LNEPS);
  F8 hr;
#pragma unroll
  for (int j = 0; j < EMB; ++j) hr.f[j] = (zr.f[j] - mu) * rs * g1[j] + be1[j];
  *(float4*)(Hout + row * EMB)     = hr.v[0];
  *(float4*)(Hout + row * EMB + 4) = hr.v[1];
  float qin[NQC], o[NQC];
#pragma unroll
  for (int j = 0; j < NQC; ++j) {
    float s = b_pre_f[j];
#pragma unroll
    for (int i = 0; i < EMB; ++i) s += hr.f[i] * w_pre_f[i * NQC + j];
    qin[j] = s;
  }
  qproj_row(qin, th_f, o);
  H8 w;
#pragma unroll
  for (int j = 0; j < NQC; ++j) w.h[j] = (_Float16)o[j];
  *(uint4*)(QFh + row * NQC) = w.u;
}

// ---------- kernel 4: fused FFN, one wave per 16-row tile, chunked over FF ----------
// ff = relu(qf @ w1 + b1) @ w2 + b2; b1 rides the A ones slot; relu intermediate goes
// through LDS with the interleaved-K packing matching w2p.

__global__ __launch_bounds__(32) void ffn_kernel(const _Float16* __restrict__ QFh,
                                                 const _Float16* __restrict__ w1h,
                                                 const _Float16* __restrict__ w2p,
                                                 const float* __restrict__ b2,
                                                 float* __restrict__ ffout) {
  const int lane = threadIdx.x;
  const int n = lane & 15;
  const int grp = lane >> 4;
  const int rt = blockIdx.x;  // ROWS/16 tiles
  const uint4 z4 = {0, 0, 0, 0};

  __shared__ __align__(16) _Float16 sQ[16 * 32];
  __shared__ __align__(16) _Float16 sT[16 * 32];
  {
    uint4* p0 = (uint4*)sQ;
    p0[lane] = z4; p0[lane + 32] = z4;
  }
  __syncthreads();
  if (lane < 16) {
    *(uint4*)(sQ + lane * 32) = *(const uint4*)(QFh + (rt * 16 + lane) * NQC);
    sQ[lane * 32 + 8] = (_Float16)1.0f;  // ones slot pairs with w1h[c][8] = b1[c]
  }
  __syncthreads();

  Frag aQ;
  aQ.u[0] = *(const uint4*)(sQ + n * 32 + grp * 8);
  aQ.u[1] = z4;

  v8f Oacc = {};
#pragma unroll 2
  for (int ffc = 0; ffc < FF; ffc += 32) {
    Frag bA, bB, bW2;
    const _Float16* a0 = w1h + (size_t)(ffc + n) * 32 + grp * 8;
    bA.u[0] = *(const uint4*)a0;  bA.u[1] = *(const uint4*)(a0 + 16);
    const _Float16* a1 = w1h + (size_t)(ffc + 16 + n) * 32 + grp * 8;
    bB.u[0] = *(const uint4*)a1;  bB.u[1] = *(const uint4*)(a1 + 16);
    const _Float16* wp = w2p + (size_t)(ffc >> 5) * 512 + n * 32 + grp * 8;
    bW2.u[0] = *(const uint4*)wp; bW2.u[1] = *(const uint4*)(wp + 16);
    if (ffc + 32 < FF) __builtin_prefetch(w1h + (size_t)(ffc + 32 + n) * 32, 0, 0);

    v8f zc = {};
    v8f T0 = __builtin_amdgcn_wmma_f32_16x16x32_f16(false, aQ.v, false, bA.v, (short)0, zc, false, false);
    v8f T1 = __builtin_amdgcn_wmma_f32_16x16x32_f16(false, aQ.v, false, bB.v, (short)0, zc, false, false);

#pragma unroll
    for (int j = 0; j < 8; ++j) {  // bias already added; relu + packed interleaved store
      int r = j + 8 * grp;
      H2 pk;
      pk.h[0] = (_Float16)fmaxf(T0[j], 0.f);   // col ffc+n    -> slot 2n
      pk.h[1] = (_Float16)fmaxf(T1[j], 0.f);   // col ffc+16+n -> slot 2n+1
      *(unsigned*)(sT + r * 32 + 2 * n) = pk.u;
    }
    __syncthreads();

    Frag aT;
    aT.u[0] = *(const uint4*)(sT + n * 32 + grp * 8);
    aT.u[1] = *(const uint4*)(sT + n * 32 + 16 + grp * 8);
    Oacc = __builtin_amdgcn_wmma_f32_16x16x32_f16(false, aT.v, false, bW2.v, (short)0, Oacc, false, false);
    __syncthreads();
  }

#pragma unroll
  for (int j = 0; j < 8; ++j) {
    if (n < EMB)
      ffout[(size_t)(rt * 16 + j + 8 * grp) * EMB + n] = Oacc[j] + b2[n];
  }
}

// ---------- kernel 5: final residual + LN2 ----------

__global__ void final_ln_kernel(const float* __restrict__ H, const float* __restrict__ ff,
                                const float* __restrict__ g2, const float* __restrict__ be2,
                                float* __restrict__ out) {
  int row = blockIdx.x * blockDim.x + threadIdx.x;
  if (row >= ROWS) return;
  F8 hr, fr, zr;
  hr.v[0] = *(const float4*)(H + row * EMB);
  hr.v[1] = *(const float4*)(H + row * EMB + 4);
  fr.v[0] = *(const float4*)(ff + row * EMB);
  fr.v[1] = *(const float4*)(ff + row * EMB + 4);
#pragma unroll
  for (int j = 0; j < EMB; ++j) zr.f[j] = hr.f[j] + fr.f[j];
  float mu = 0.f;
#pragma unroll
  for (int j = 0; j < EMB; ++j) mu += zr.f[j];
  mu *= (1.f / EMB);
  float var = 0.f;
#pragma unroll
  for (int j = 0; j < EMB; ++j) { float d = zr.f[j] - mu; var += d * d; }
  var *= (1.f / EMB);
  float rs = rsqrtf(var + LNEPS);
  F8 orow;
#pragma unroll
  for (int j = 0; j < EMB; ++j) orow.f[j] = (zr.f[j] - mu) * rs * g2[j] + be2[j];
  *(float4*)(out + row * EMB)     = orow.v[0];
  *(float4*)(out + row * EMB + 4) = orow.v[1];
}

// ---------- launch ----------

extern "C" void kernel_launch(void* const* d_in, const int* in_sizes, int n_in,
                              void* d_out, int out_size, void* d_ws, size_t ws_size,
                              hipStream_t stream) {
  const float* x       = (const float*)d_in[0];
  const float* w_pre   = (const float*)d_in[1];
  const float* b_pre   = (const float*)d_in[2];
  const float* th_q    = (const float*)d_in[3];
  const float* th_k    = (const float*)d_in[4];
  const float* th_v    = (const float*)d_in[5];
  const float* w_comb  = (const float*)d_in[6];
  const float* b_comb  = (const float*)d_in[7];
  const float* g1      = (const float*)d_in[8];
  const float* be1     = (const float*)d_in[9];
  const float* g2      = (const float*)d_in[10];
  const float* be2     = (const float*)d_in[11];
  const float* w_pre_f = (const float*)d_in[12];
  const float* b_pre_f = (const float*)d_in[13];
  const float* th_f    = (const float*)d_in[14];
  const float* w1      = (const float*)d_in[15];
  const float* b1      = (const float*)d_in[16];
  const float* w2      = (const float*)d_in[17];
  const float* b2      = (const float*)d_in[18];
  (void)in_sizes; (void)n_in; (void)out_size; (void)ws_size;

  char* base = (char*)d_ws;
  _Float16* Qh  = (_Float16*)(base + 0);        // 256 KB
  _Float16* Kh  = (_Float16*)(base + 262144);   // 256 KB
  _Float16* Vh  = (_Float16*)(base + 524288);   // 256 KB
  float*    attn= (float*)   (base + 786432);   // 512 KB
  float*    H   = (float*)   (base + 1310720);  // 512 KB
  _Float16* QFh = (_Float16*)(base + 1835008);  // 256 KB
  float*    FFo = (float*)   (base + 2097152);  // 512 KB
  _Float16* w1h = (_Float16*)(base + 2621440);  // 256 KB
  _Float16* w2p = (_Float16*)(base + 2883584);  // 128 KB
  float* out = (float*)d_out;

  qkv_kernel<<<ROWS / 256, 256, 0, stream>>>(x, w_pre, b_pre, th_q, th_k, th_v, Qh, Kh, Vh);
  pack_w_kernel<<<FF / 256, 256, 0, stream>>>(w1, b1, w2, w1h, w2p);
  attn_flash_kernel<<<NB * (SEQ / 16), 32, 0, stream>>>(Qh, Kh, Vh, attn);
  postattn_kernel<<<ROWS / 256, 256, 0, stream>>>(x, attn, w_comb, b_comb, g1, be1,
                                                  w_pre_f, b_pre_f, th_f, H, QFh);
  ffn_kernel<<<ROWS / 16, 32, 0, stream>>>(QFh, w1h, w2p, b2, FFo);
  final_ln_kernel<<<ROWS / 256, 256, 0, stream>>>(H, FFo, g2, be2, out);
}
